// GATLayer_3075196584516
// MI455X (gfx1250) — matrix operand order repaired and checked by reading, not compiled, not run on hardware
//
#include <hip/hip_runtime.h>
#include <hip/hip_bf16.h>

typedef __bf16 v16bf __attribute__((ext_vector_type(16)));
typedef float  v8f   __attribute__((ext_vector_type(8)));

#define IN_DIM 256
#define HC 256      // HEADS * OUT_DIM
#define ODIM 64
#define HEADS 4
#define NEG_SLOPE 0.2f
#define KTILES 8    // 256 / 32
#define NTILES 16   // 256 / 16

// ---------- helpers ----------
static __device__ __forceinline__ __bf16 f2bf(float f) {
    union { float f; unsigned u; } a; a.f = f;
    unsigned r = a.u + 0x7FFFu + ((a.u >> 16) & 1u);   // round-to-nearest-even
    unsigned short h = (unsigned short)(r >> 16);
    __bf16 b;
    __builtin_memcpy(&b, &h, 2);
    return b;
}

static __device__ __forceinline__ void atomicMaxF32(float* addr, float val) {
    if (val >= 0.0f) atomicMax((int*)addr, __float_as_int(val));
    else             atomicMin((unsigned int*)addr, __float_as_uint(val));
}

// A fragment loader: ISA 16-bit A layout (16x32). Per lane, K values are two
// contiguous 8-float runs -> 4x global_load_b128, then cvt to bf16.
static __device__ __forceinline__ v16bf loadAfrag(const float* __restrict__ p) {
    const float4* v0 = (const float4*)p;          // K base .. base+7
    const float4* v1 = (const float4*)(p + 16);   // K base+16 .. base+23
    float4 r0 = v0[0], r1 = v0[1], r2 = v1[0], r3 = v1[1];
    v16bf a;
    a[0]  = f2bf(r0.x); a[1]  = f2bf(r0.y); a[2]  = f2bf(r0.z); a[3]  = f2bf(r0.w);
    a[4]  = f2bf(r1.x); a[5]  = f2bf(r1.y); a[6]  = f2bf(r1.z); a[7]  = f2bf(r1.w);
    a[8]  = f2bf(r2.x); a[9]  = f2bf(r2.y); a[10] = f2bf(r2.z); a[11] = f2bf(r2.w);
    a[12] = f2bf(r3.x); a[13] = f2bf(r3.y); a[14] = f2bf(r3.z); a[15] = f2bf(r3.w);
    return a;
}

// ---------- 0: pack W (fp32, [K=256][N=256] row-major) into WMMA B-fragment layout ----------
// tile = kTile*16 + nTile; per (tile, lane): 16 bf16 contiguous (32B) so GEMM loads 1 v16bf.
// ISA 16-bit B layout (32x16): element e -> K = kTile*32 + (lane>=16?8:0) + e + (e>=8?8:0), col = nTile*16 + (lane&15)
__global__ void k_packB(const float* __restrict__ W, __bf16* __restrict__ Bpk) {
    int t = blockIdx.x * blockDim.x + threadIdx.x;     // tile*32 + lane
    if (t >= KTILES * NTILES * 32) return;
    int lane  = t & 31;
    int tile  = t >> 5;
    int kTile = tile >> 4;
    int nTile = tile & 15;
    int k0  = kTile * 32 + ((lane >= 16) ? 8 : 0);
    int col = nTile * 16 + (lane & 15);
    __bf16* dst = Bpk + (size_t)t * 16;
#pragma unroll
    for (int e = 0; e < 16; ++e)
        dst[e] = f2bf(W[(size_t)(k0 + e + ((e >= 8) ? 8 : 0)) * HC + col]);
}

// ---------- 1: H = X @ W via WMMA bf16 ----------
// grid.x = ceil(N/32); block = 128 (4 waves).
// Wave w: rows [32*blk, +32) (2 M-tiles), cols [64*w, +64) (4 N-tiles) -> 8 wmma per k-step,
// reusing the 4 B fragments across both M-tiles.
__global__ __launch_bounds__(128) void k_gemm_wmma(const float* __restrict__ X,
                                                   const v16bf* __restrict__ Bpk,
                                                   float* __restrict__ H, int N) {
    const int wave = threadIdx.x >> 5;
    const int lane = threadIdx.x & 31;
    const int row0 = blockIdx.x * 32;
    const int nBase = wave * 64;
    const int nTile0 = wave * 4;

    int aRow0 = row0 + (lane & 15);
    int aRow1 = row0 + 16 + (lane & 15);
    if (aRow0 >= N) aRow0 = N - 1;               // clamp; OOB rows not stored
    if (aRow1 >= N) aRow1 = N - 1;
    const int aKbase = (lane >= 16) ? 8 : 0;     // ISA 16-bit A layout
    const float* __restrict__ xrow0 = X + (size_t)aRow0 * IN_DIM + aKbase;
    const float* __restrict__ xrow1 = X + (size_t)aRow1 * IN_DIM + aKbase;
    const int bCol0 = lane & 15;

    v8f acc[8] = {};     // acc[mi*4 + nt]
    for (int k0 = 0; k0 < IN_DIM; k0 += 32) {
        if (k0 + 32 < IN_DIM) {
            __builtin_prefetch(xrow0 + k0 + 32, 0, 3);   // global_prefetch_b8 (near)
            __builtin_prefetch(xrow1 + k0 + 32, 0, 3);
        }
        // batch all B loads first so wmma can overlap with outstanding loads
        const int kTile = k0 >> 5;
        v16bf b[4];
#pragma unroll
        for (int nt = 0; nt < 4; ++nt)
            b[nt] = Bpk[(size_t)((kTile * NTILES + nTile0 + nt) * 32 + lane)];

        v16bf a0 = loadAfrag(xrow0 + k0);
        v16bf a1 = loadAfrag(xrow1 + k0);
#pragma unroll
        for (int nt = 0; nt < 4; ++nt)
            acc[nt] = __builtin_amdgcn_wmma_f32_16x16x32_bf16(
                false, a0, false, b[nt], (short)0, acc[nt], false, false);
#pragma unroll
        for (int nt = 0; nt < 4; ++nt)
            acc[4 + nt] = __builtin_amdgcn_wmma_f32_16x16x32_bf16(
                false, a1, false, b[nt], (short)0, acc[4 + nt], false, false);
    }

    // C/D layout: VGPR r -> M = r + (lane>=16 ? 8 : 0), N = lane&15
    const int rOff = (lane >= 16) ? 8 : 0;
    if (row0 + 32 <= N) {          // fast path: no per-row guards
#pragma unroll
        for (int mi = 0; mi < 2; ++mi) {
#pragma unroll
            for (int nt = 0; nt < 4; ++nt) {
                const int col = nBase + nt * 16 + bCol0;
                float* hp = H + (size_t)(row0 + mi * 16 + rOff) * HC + col;
#pragma unroll
                for (int r = 0; r < 8; ++r)
                    hp[(size_t)r * HC] = acc[mi * 4 + nt][r];
            }
        }
    } else {
#pragma unroll
        for (int mi = 0; mi < 2; ++mi) {
#pragma unroll
            for (int nt = 0; nt < 4; ++nt) {
                const int col = nBase + nt * 16 + bCol0;
#pragma unroll
                for (int r = 0; r < 8; ++r) {
                    int row = row0 + mi * 16 + rOff + r;
                    if (row < N) H[(size_t)row * HC + col] = acc[mi * 4 + nt][r];
                }
            }
        }
    }
}

// ---------- 2: a_src / a_dst per (node, head) ----------
__global__ void k_attn_dots(const float* __restrict__ H,
                            const float* __restrict__ attS,
                            const float* __restrict__ attD,
                            float* __restrict__ aS, float* __restrict__ aD, int N) {
    int idx = blockIdx.x * blockDim.x + threadIdx.x;
    if (idx >= N * HEADS) return;
    int n = idx >> 2, h = idx & 3;
    const float4* hp = (const float4*)(H + (size_t)n * HC + h * ODIM);
    const float4* s  = (const float4*)(attS + h * ODIM);
    const float4* d  = (const float4*)(attD + h * ODIM);
    float accS = 0.f, accD = 0.f;
#pragma unroll
    for (int c = 0; c < ODIM / 4; ++c) {
        float4 hv = hp[c], sv = s[c], dv = d[c];
        accS += hv.x * sv.x + hv.y * sv.y + hv.z * sv.z + hv.w * sv.w;
        accD += hv.x * dv.x + hv.y * dv.y + hv.z * dv.z + hv.w * dv.w;
    }
    aS[idx] = accS;
    aD[idx] = accD;
}

// ---------- 3: init m=-1e30, denom=0, out=0 ----------
__global__ void k_init(float* __restrict__ m, float* __restrict__ denom,
                       float* __restrict__ out, int N) {
    int idx = blockIdx.x * blockDim.x + threadIdx.x;
    if (idx < N * HC) out[idx] = 0.0f;
    if (idx < N * HEADS) { m[idx] = -1.0e30f; denom[idx] = 0.0f; }
}

// ---------- 4: edge logits + leaky relu + segment max ----------
__global__ void k_edge_logits(const int* __restrict__ ei, int E, int N,
                              const float* __restrict__ aS, const float* __restrict__ aD,
                              float* __restrict__ eBuf, float* __restrict__ m) {
    int idx = blockIdx.x * blockDim.x + threadIdx.x;
    int total = (E + N) * HEADS;
    if (idx >= total) return;
    int eId = idx >> 2, h = idx & 3;
    int s, d;
    if (eId < E) { s = ei[eId]; d = ei[E + eId]; }
    else         { s = eId - E; d = s; }           // self loop
    float v = aS[s * HEADS + h] + aD[d * HEADS + h];
    v = (v > 0.0f) ? v : NEG_SLOPE * v;
    eBuf[idx] = v;
    atomicMaxF32(&m[d * HEADS + h], v);
}

// ---------- 5: p = exp(e - m[dst]); denom += p ----------
__global__ void k_edge_exp(const int* __restrict__ ei, int E, int N,
                           const float* __restrict__ m,
                           float* __restrict__ eBuf, float* __restrict__ denom) {
    int idx = blockIdx.x * blockDim.x + threadIdx.x;
    int total = (E + N) * HEADS;
    if (idx >= total) return;
    int eId = idx >> 2, h = idx & 3;
    int d = (eId < E) ? ei[E + eId] : (eId - E);
    float p = __expf(eBuf[idx] - m[d * HEADS + h]);
    eBuf[idx] = p;
    atomicAdd(&denom[d * HEADS + h], p);
}

// ---------- 6: out[dst] += alpha * h[src]  (one wave per edge) ----------
__global__ __launch_bounds__(256) void k_edge_aggregate(const int* __restrict__ ei, int E, int N,
                                                        const float* __restrict__ p,
                                                        const float* __restrict__ denom,
                                                        const float* __restrict__ H,
                                                        float* __restrict__ out) {
    int gtid = blockIdx.x * blockDim.x + threadIdx.x;
    int waveId = gtid >> 5;
    int lane = gtid & 31;
    if (waveId >= E + N) return;
    int s, d;
    if (waveId < E) { s = ei[waveId]; d = ei[E + waveId]; }
    else            { s = waveId - E; d = s; }
    const float* hs = H + (size_t)s * HC;
    float* od = out + (size_t)d * HC;
    const float* pe = p + (size_t)waveId * HEADS;
    const float* de = denom + (size_t)d * HEADS;
    float alpha[HEADS];
#pragma unroll
    for (int h = 0; h < HEADS; ++h) alpha[h] = pe[h] / (de[h] + 1e-16f);
#pragma unroll
    for (int j = 0; j < 8; ++j) {
        int c = lane + j * 32;
        atomicAdd(&od[c], alpha[c >> 6] * hs[c]);
    }
}

// ---------- 7: out = relu(out + bias) ----------
__global__ void k_finalize(float* __restrict__ out, const float* __restrict__ bias, int N) {
    int idx = blockIdx.x * blockDim.x + threadIdx.x;
    if (idx >= N * HC) return;
    float v = out[idx] + bias[idx & (HC - 1)];
    out[idx] = (v > 0.0f) ? v : 0.0f;
}

extern "C" void kernel_launch(void* const* d_in, const int* in_sizes, int n_in,
                              void* d_out, int out_size, void* d_ws, size_t ws_size,
                              hipStream_t stream) {
    const float* x    = (const float*)d_in[0];
    const int*   ei   = (const int*)d_in[1];
    const float* W    = (const float*)d_in[2];
    const float* attS = (const float*)d_in[3];
    const float* attD = (const float*)d_in[4];
    const float* bias = (const float*)d_in[5];
    float* out = (float*)d_out;

    const int N = in_sizes[0] / IN_DIM;     // 50000
    const int E = in_sizes[1] / 2;          // 800000

    // workspace carve-up (all offsets 256B-aligned)
    char* ws = (char*)d_ws;
    size_t off = 0;
    __bf16* Bpk  = (__bf16*)(ws + off); off += (size_t)KTILES * NTILES * 32 * 16 * 2; // 128 KB
    float*  H    = (float*)(ws + off);  off += (size_t)N * HC * 4;                    // 51.2 MB
    float*  aS   = (float*)(ws + off);  off += (size_t)N * HEADS * 4;
    float*  aD   = (float*)(ws + off);  off += (size_t)N * HEADS * 4;
    float*  m    = (float*)(ws + off);  off += (size_t)N * HEADS * 4;
    float*  den  = (float*)(ws + off);  off += (size_t)N * HEADS * 4;
    float*  eBuf = (float*)(ws + off);  off += (size_t)(E + N) * HEADS * 4;           // 13.6 MB

    k_packB<<<(KTILES * NTILES * 32 + 255) / 256, 256, 0, stream>>>(W, Bpk);

    k_gemm_wmma<<<(N + 31) / 32, 128, 0, stream>>>(x, (const v16bf*)Bpk, H, N);

    k_attn_dots<<<(N * HEADS + 255) / 256, 256, 0, stream>>>(H, attS, attD, aS, aD, N);

    k_init<<<((size_t)N * HC + 255) / 256, 256, 0, stream>>>(m, den, out, N);

    int totalEH = (E + N) * HEADS;
    k_edge_logits<<<(totalEH + 255) / 256, 256, 0, stream>>>(ei, E, N, aS, aD, eBuf, m);
    k_edge_exp<<<(totalEH + 255) / 256, 256, 0, stream>>>(ei, E, N, m, eBuf, den);

    size_t aggThreads = (size_t)(E + N) * 32;
    k_edge_aggregate<<<(aggThreads + 255) / 256, 256, 0, stream>>>(ei, E, N, eBuf, den, H, out);

    k_finalize<<<((size_t)N * HC + 255) / 256, 256, 0, stream>>>(out, bias, N);
}
